// NeighborTransferModel_3805341024596
// MI455X (gfx1250) — matrix operand order repaired and checked by reading, not compiled, not run on hardware
//
#include <hip/hip_runtime.h>
#include <hip/hip_bf16.h>

typedef __attribute__((ext_vector_type(16))) __bf16 v16bf;
typedef __attribute__((ext_vector_type(8)))  float  v8f;
typedef __hip_bfloat16 bf16;

#define N_ 1024
#define K_ 32
#define P_ 1024
#define D_ 512
#define H_ 256
#define R_ (N_*K_)   // 32768 rows
#define KC 1536      // folded concat K-dim (P + D)
#define CW 512       // concat hidden width (weight-net 256 | delta-net 256)

// ---------------- prep: fold + transpose weights to bf16 [col][k] ----------------
__global__ void prep_weights(const float* __restrict__ w1, const float* __restrict__ wb1,
                             const float* __restrict__ d1, const float* __restrict__ db1,
                             const float* __restrict__ w2, const float* __restrict__ wb2,
                             const float* __restrict__ d2, const float* __restrict__ db2,
                             const float* __restrict__ be1,
                             bf16* __restrict__ Bcat, bf16* __restrict__ Acat,
                             bf16* __restrict__ BE1cat, bf16* __restrict__ W2cat,
                             float* __restrict__ Ws, float* __restrict__ b1cat,
                             float* __restrict__ b2cat)
{
  int idx = blockIdx.x * blockDim.x + threadIdx.x;
  if (idx >= CW * KC) return;
  int j = idx / KC;      // output column 0..511 (0..255 weight-net, 256..511 delta-net)
  int kk = idx % KC;     // folded input feature 0..1535 (0..1023 protein, 1024..1535 drug)
  const float* W = (j < H_) ? w1 : d1;
  int h = j & (H_ - 1);
  float bq, aq;
  if (kk < P_) {
    int p = kk;
    bq = W[(P_ + p) * H_ + h]     - W[(2 * P_ + p) * H_ + h];   // np coeff: Wb - Wc
    aq = W[p * H_ + h]            + W[(2 * P_ + p) * H_ + h];   // qp coeff: Wa + Wc
  } else {
    int dd = kk - P_;
    bq = W[(3 * P_ + D_ + dd) * H_ + h]     - W[(3 * P_ + 2 * D_ + dd) * H_ + h]; // nd: We - Wf
    aq = W[(3 * P_ + dd) * H_ + h]          + W[(3 * P_ + 2 * D_ + dd) * H_ + h]; // qd: Wd + Wf
  }
  Bcat[j * KC + kk] = __float2bfloat16(bq);
  Acat[j * KC + kk] = __float2bfloat16(aq);
  if (j < H_) BE1cat[j * KC + kk] = __float2bfloat16(be1[kk * H_ + j]);
  if (j < H_ && kk < CW) {   // block-diagonal fused layer-2 weights [out 256][k 512]
    float v = 0.f;
    if (j < 128 && kk < 256)        v = w2[kk * 128 + j];
    else if (j >= 128 && kk >= 256) v = d2[(kk - 256) * 128 + (j - 128)];
    W2cat[j * CW + kk] = __float2bfloat16(v);
  }
  if (kk < 4) Ws[kk * CW + j] = (j < H_) ? w1[(3*P_+3*D_+kk)*H_ + j] : d1[(3*P_+3*D_+kk)*H_ + (j-H_)];
  if (kk == 4) b1cat[j] = (j < H_) ? wb1[j] : db1[j - H_];
  if (kk == 5 && j < H_) b2cat[j] = (j < 128) ? wb2[j] : db2[j - 128];
}

__global__ void prep_qcat(const float* __restrict__ qp, const float* __restrict__ qd,
                          bf16* __restrict__ qcat)
{
  int idx = blockIdx.x * blockDim.x + threadIdx.x;
  if (idx >= N_ * KC) return;
  int n = idx / KC, kk = idx % KC;
  float v = (kk < P_) ? qp[n * P_ + kk] : qd[n * D_ + (kk - P_)];
  qcat[idx] = __float2bfloat16(v);
}

// per-row scalars: [aff, ppr, trust, cosine_sim(qd, nd)]
__global__ void prep_scal(const float* __restrict__ aff, const float* __restrict__ ppr,
                          const float* __restrict__ tru, const float* __restrict__ qd,
                          const float* __restrict__ nd, float* __restrict__ scal)
{
  int r = blockIdx.x;
  int lane = threadIdx.x;
  int n = r >> 5;
  float dot = 0.f, nq = 0.f, nn = 0.f;
  for (int j = lane; j < D_; j += 32) {
    float a = qd[n * D_ + j], b = nd[(size_t)r * D_ + j];
    dot += a * b; nq += a * a; nn += b * b;
  }
  for (int o = 16; o; o >>= 1) {
    dot += __shfl_xor(dot, o, 32);
    nq  += __shfl_xor(nq,  o, 32);
    nn  += __shfl_xor(nn,  o, 32);
  }
  if (lane == 0) {
    float cs = dot / (fmaxf(sqrtf(nq), 1e-8f) * fmaxf(sqrtf(nn), 1e-8f));
    scal[r * 4 + 0] = aff[r]; scal[r * 4 + 1] = ppr[r];
    scal[r * 4 + 2] = tru[r]; scal[r * 4 + 3] = cs;
  }
}

__device__ __forceinline__ void async_copy_b128(unsigned lds_addr, const void* gaddr) {
  // GLOBAL_LOAD_ASYNC_TO_LDS_B128: memory -> LDS, tracked by ASYNCcnt (no VGPR data path)
  asm volatile("global_load_async_to_lds_b128 %0, %1, off"
               :: "v"(lds_addr), "v"((unsigned long long)(uintptr_t)gaddr)
               : "memory");
}
__device__ __forceinline__ void wait_async0() {
  asm volatile("s_wait_asynccnt 0x0" ::: "memory");
}

// ---------------- generic 128x128-tile bf16 WMMA GEMM ----------------
// Double-buffered LDS; B tile (and A tile when already bf16) staged with
// global_load_async_to_lds_b128; dualA path software-pipelines f32->bf16 via regs.
// modes: 0 = f32 out + bias(e0);  1 = +Qterm(e0) + scal(e2)·Ws(e1), relu, bf16 out;
//        2 = +bias(e0), relu, bf16 out
__global__ __launch_bounds__(256)
void gemm_wmma(int dualA, const bf16* __restrict__ Abf,
               const float* __restrict__ Anp, const float* __restrict__ And,
               int Kdim, const bf16* __restrict__ Bc, int mode,
               const float* __restrict__ e0, const float* __restrict__ e1,
               const float* __restrict__ e2, void* __restrict__ outp, int ldc)
{
  __shared__ __align__(16) bf16 Al[2][128][32];
  __shared__ __align__(16) bf16 Bl[2][128][32];
  int tid = threadIdx.x, lane = tid & 31, wave = tid >> 5;
  int wrow = wave >> 2;          // 0..1 : 64-row half
  int wcol = wave & 3;           // 0..3 : 32-col strip
  int rowBase = blockIdx.x * 128, colBase = blockIdx.y * 128;
  int kg = lane >> 4, ml = lane & 15;

  // each thread owns 2 16-byte chunks of each 128x32 tile (512 chunks total)
  int c0 = tid, c1 = tid + 256;
  int r0 = c0 >> 2, cc0 = c0 & 3;
  int r1 = c1 >> 2, cc1 = c1 & 3;

  auto issueB = [&](int kb, int buf) {
    int k0 = kb * 32;
    async_copy_b128((unsigned)(uintptr_t)&Bl[buf][r0][cc0 * 8],
                    Bc + (size_t)(colBase + r0) * Kdim + k0 + cc0 * 8);
    async_copy_b128((unsigned)(uintptr_t)&Bl[buf][r1][cc1 * 8],
                    Bc + (size_t)(colBase + r1) * Kdim + k0 + cc1 * 8);
  };
  auto issueA = [&](int kb, int buf) {
    int k0 = kb * 32;
    async_copy_b128((unsigned)(uintptr_t)&Al[buf][r0][cc0 * 8],
                    Abf + (size_t)(rowBase + r0) * Kdim + k0 + cc0 * 8);
    async_copy_b128((unsigned)(uintptr_t)&Al[buf][r1][cc1 * 8],
                    Abf + (size_t)(rowBase + r1) * Kdim + k0 + cc1 * 8);
  };
  auto aptr = [&](int r, int k) -> const float* {
    return (k < P_) ? (Anp + (size_t)(rowBase + r) * P_ + k)
                    : (And + (size_t)(rowBase + r) * D_ + (k - P_));
  };
  auto cvst = [&](float4 a, float4 b, bf16* dst) {
    dst[0] = __float2bfloat16(a.x); dst[1] = __float2bfloat16(a.y);
    dst[2] = __float2bfloat16(a.z); dst[3] = __float2bfloat16(a.w);
    dst[4] = __float2bfloat16(b.x); dst[5] = __float2bfloat16(b.y);
    dst[6] = __float2bfloat16(b.z); dst[7] = __float2bfloat16(b.w);
  };

  v8f acc[4][2];
  for (int mt = 0; mt < 4; ++mt)
    for (int nt = 0; nt < 2; ++nt)
      for (int v = 0; v < 8; ++v) acc[mt][nt][v] = 0.f;

  union FragU { uint4 u[2]; v16bf v; };
  int nk = Kdim / 32;

  // ---- prologue: stage tile 0 into buffer 0 ----
  if (dualA) {
    const float* s0 = aptr(r0, cc0 * 8);
    const float* s1 = aptr(r1, cc1 * 8);
    cvst(((const float4*)s0)[0], ((const float4*)s0)[1], &Al[0][r0][cc0 * 8]);
    cvst(((const float4*)s1)[0], ((const float4*)s1)[1], &Al[0][r1][cc1 * 8]);
  } else {
    issueA(0, 0);
  }
  issueB(0, 0);
  wait_async0();
  __syncthreads();

  for (int kb = 0; kb < nk; ++kb) {
    int cur = kb & 1, nxt = cur ^ 1;
    bool more = (kb + 1) < nk;
    float4 pf[4];
    if (more) {
      if (dualA) {  // prefetch next A tile into registers (hidden under WMMA below)
        int k = (kb + 1) * 32;
        const float* s0 = aptr(r0, k + cc0 * 8);
        const float* s1 = aptr(r1, k + cc1 * 8);
        pf[0] = ((const float4*)s0)[0]; pf[1] = ((const float4*)s0)[1];
        pf[2] = ((const float4*)s1)[0]; pf[3] = ((const float4*)s1)[1];
      } else {
        issueA(kb + 1, nxt);
      }
      issueB(kb + 1, nxt);
    }

    // ---- compute on buffer `cur` ----
    v16bf bfrag[2];
    for (int nt = 0; nt < 2; ++nt) {   // B 32x16: lanes0-15 K0..15, lanes16-31 K16..31
      FragU f;
      int col = wcol * 32 + nt * 16 + ml;
      f.u[0] = *(const uint4*)&Bl[cur][col][kg * 16];
      f.u[1] = *(const uint4*)&Bl[cur][col][kg * 16 + 8];
      bfrag[nt] = f.v;
    }
    for (int mt = 0; mt < 4; ++mt) {   // A 16x32: lanes0-15 K0..15, lanes16-31 K16..31
      FragU f;
      int row = wrow * 64 + mt * 16 + ml;
      f.u[0] = *(const uint4*)&Al[cur][row][kg * 16];
      f.u[1] = *(const uint4*)&Al[cur][row][kg * 16 + 8];
      v16bf afrag = f.v;
      for (int nt = 0; nt < 2; ++nt)
        acc[mt][nt] = __builtin_amdgcn_wmma_f32_16x16x32_bf16(
            false, afrag, false, bfrag[nt], (short)0, acc[mt][nt], false, false);
    }

    if (more && dualA) {  // convert prefetched A regs into the other buffer
      cvst(pf[0], pf[1], &Al[nxt][r0][cc0 * 8]);
      cvst(pf[2], pf[3], &Al[nxt][r1][cc1 * 8]);
    }
    wait_async0();       // our async copies for `nxt` landed in LDS
    __syncthreads();     // publish to all waves (also waits our ds stores)
  }

  // epilogue; C/D layout: VGPR v, lanes0-15: M=v,N=lane; lanes16-31: M=v+8,N=lane-16
  for (int mt = 0; mt < 4; ++mt)
    for (int nt = 0; nt < 2; ++nt)
      for (int v = 0; v < 8; ++v) {
        int row = rowBase + wrow * 64 + mt * 16 + kg * 8 + v;
        int col = colBase + wcol * 32 + nt * 16 + ml;
        float t = acc[mt][nt][v];
        if (mode == 0) {
          ((float*)outp)[(size_t)row * ldc + col] = t + e0[col];
        } else if (mode == 1) {
          int n = row >> 5;
          t += e0[(size_t)n * CW + col];
          t += e2[row * 4 + 0] * e1[0 * CW + col] + e2[row * 4 + 1] * e1[1 * CW + col]
             + e2[row * 4 + 2] * e1[2 * CW + col] + e2[row * 4 + 3] * e1[3 * CW + col];
          ((bf16*)outp)[(size_t)row * ldc + col] = __float2bfloat16(fmaxf(t, 0.f));
        } else {
          ((bf16*)outp)[(size_t)row * ldc + col] = __float2bfloat16(fmaxf(t + e0[col], 0.f));
        }
      }
}

// ---------------- finalize: layer-3 dots, masked softmax, prior, sigma ----------------
__global__ void finalize(const bf16* __restrict__ h2, const bf16* __restrict__ hbe,
                         const float* __restrict__ w3, const float* __restrict__ wb3,
                         const float* __restrict__ d3, const float* __restrict__ db3,
                         const float* __restrict__ be2, const float* __restrict__ beb2,
                         const float* __restrict__ log_sigma,
                         const float* __restrict__ aff, const unsigned char* __restrict__ mask,
                         float* __restrict__ out)
{
  int n = blockIdx.x, lane = threadIdx.x;   // lane == neighbor index k
  int r = n * 32 + lane;
  const bf16* hr = h2 + (size_t)r * 256;
  float lg = 0.f, dl = 0.f;
  for (int j = 0; j < 128; ++j) lg += __bfloat162float(hr[j]) * w3[j];
  for (int j = 0; j < 128; ++j) dl += __bfloat162float(hr[128 + j]) * d3[j];
  lg += wb3[0]; dl += db3[0];
  int m = mask[r] ? 1 : 0;
  float l = m ? lg : -1e9f;
  float mx = l;
  for (int o = 16; o; o >>= 1) mx = fmaxf(mx, __shfl_xor(mx, o, 32));
  float e = expf(l - mx);
  float s = e;
  for (int o = 16; o; o >>= 1) s += __shfl_xor(s, o, 32);
  float w = (e / s) * (float)m;
  float norm = w;
  for (int o = 16; o; o >>= 1) norm += __shfl_xor(norm, o, 32);
  w = (norm > 0.f) ? w / fmaxf(norm, 1e-8f) : 0.f;
  float tr = w * aff[r];
  float co = w * dl;
  for (int o = 16; o; o >>= 1) { tr += __shfl_xor(tr, o, 32); co += __shfl_xor(co, o, 32); }
  int anym = m;
  for (int o = 16; o; o >>= 1) anym |= __shfl_xor(anym, o, 32);
  float pr = 0.f;
  for (int i = 0; i < 8; ++i) {
    int j = lane * 8 + i;
    pr += __bfloat162float(hbe[(size_t)n * 256 + j]) * be2[j];
  }
  for (int o = 16; o; o >>= 1) pr += __shfl_xor(pr, o, 32);
  pr += beb2[0] + 6.5f;  // GLOBAL_MEAN_AFFINITY
  if (lane == 0) {
    float mu = anym ? (tr + co) : pr;
    float sigma = log1pf(expf(log_sigma[0])) + (anym ? 0.f : 0.15f) + 1e-4f;
    out[n] = mu;
    out[N_ + n] = sigma;
  }
}

extern "C" void kernel_launch(void* const* d_in, const int* in_sizes, int n_in,
                              void* d_out, int out_size, void* d_ws, size_t ws_size,
                              hipStream_t stream) {
  const float* np  = (const float*)d_in[0];
  const float* nd  = (const float*)d_in[1];
  const float* aff = (const float*)d_in[2];
  const float* ppr = (const float*)d_in[3];
  const float* tru = (const float*)d_in[4];
  const unsigned char* mask = (const unsigned char*)d_in[5];
  const float* qp  = (const float*)d_in[6];
  const float* qd  = (const float*)d_in[7];
  const float* w1  = (const float*)d_in[8];
  const float* wb1 = (const float*)d_in[9];
  const float* w2  = (const float*)d_in[10];
  const float* wb2 = (const float*)d_in[11];
  const float* w3  = (const float*)d_in[12];
  const float* wb3 = (const float*)d_in[13];
  const float* d1  = (const float*)d_in[14];
  const float* db1 = (const float*)d_in[15];
  const float* d2  = (const float*)d_in[16];
  const float* db2 = (const float*)d_in[17];
  const float* d3  = (const float*)d_in[18];
  const float* db3 = (const float*)d_in[19];
  const float* be1 = (const float*)d_in[20];
  const float* beb1= (const float*)d_in[21];
  const float* be2 = (const float*)d_in[22];
  const float* beb2= (const float*)d_in[23];
  const float* lsg = (const float*)d_in[24];

  char* base = (char*)d_ws;
  size_t off = 0;
  auto alloc = [&](size_t bytes) {
    void* p = base + off;
    off = (off + bytes + 255) & ~(size_t)255;
    return p;
  };
  bf16*  Bcat    = (bf16*) alloc((size_t)CW * KC * 2);      // folded np/nd coeffs
  bf16*  Acat    = (bf16*) alloc((size_t)CW * KC * 2);      // folded qp/qd coeffs
  bf16*  BE1cat  = (bf16*) alloc((size_t)H_ * KC * 2);      // binding encoder L1, transposed
  bf16*  W2cat   = (bf16*) alloc((size_t)H_ * CW * 2);      // block-diag fused layer2
  bf16*  qcat    = (bf16*) alloc((size_t)N_ * KC * 2);      // [qp|qd] bf16
  float* Ws      = (float*)alloc((size_t)4 * CW * 4);       // scalar-row weights
  float* b1cat   = (float*)alloc((size_t)CW * 4);
  float* b2cat   = (float*)alloc((size_t)H_ * 4);
  float* Qterm   = (float*)alloc((size_t)N_ * CW * 4);      // query contribution + b1
  float* scal    = (float*)alloc((size_t)R_ * 4 * 4);       // per-row scalars
  bf16*  hidden1 = (bf16*) alloc((size_t)R_ * CW * 2);      // relu layer1, both nets
  bf16*  h2      = (bf16*) alloc((size_t)R_ * H_ * 2);      // relu layer2, both nets
  bf16*  hbe     = (bf16*) alloc((size_t)N_ * H_ * 2);      // binding encoder hidden
  (void)ws_size; (void)in_sizes; (void)n_in; (void)out_size;

  prep_weights<<<(CW * KC + 255) / 256, 256, 0, stream>>>(
      w1, wb1, d1, db1, w2, wb2, d2, db2, be1,
      Bcat, Acat, BE1cat, W2cat, Ws, b1cat, b2cat);
  prep_qcat<<<(N_ * KC + 255) / 256, 256, 0, stream>>>(qp, qd, qcat);
  prep_scal<<<R_, 32, 0, stream>>>(aff, ppr, tru, qd, nd, scal);

  // Qterm[n, j] = qcat @ Acat + b1cat   (1024 x 512)
  gemm_wmma<<<dim3(N_ / 128, CW / 128), 256, 0, stream>>>(
      0, qcat, nullptr, nullptr, KC, Acat, 0, b1cat, nullptr, nullptr, Qterm, CW);
  // binding encoder hidden: relu(qcat @ BE1 + beb1)   (1024 x 256)
  gemm_wmma<<<dim3(N_ / 128, H_ / 128), 256, 0, stream>>>(
      0, qcat, nullptr, nullptr, KC, BE1cat, 2, beb1, nullptr, nullptr, hbe, H_);
  // layer1: relu([np|nd] @ Bcat + Qterm + scal·Ws)   (32768 x 512)  — dominant GEMM
  gemm_wmma<<<dim3(R_ / 128, CW / 128), 256, 0, stream>>>(
      1, nullptr, np, nd, KC, Bcat, 1, Qterm, Ws, scal, hidden1, CW);
  // layer2: relu(hidden1 @ W2cat + b2cat)   (32768 x 256)
  gemm_wmma<<<dim3(R_ / 128, H_ / 128), 256, 0, stream>>>(
      0, hidden1, nullptr, nullptr, CW, W2cat, 2, b2cat, nullptr, nullptr, h2, H_);

  finalize<<<N_, 32, 0, stream>>>(h2, hbe, w3, wb3, d3, db3, be2, beb2, lsg,
                                  aff, mask, (float*)d_out);
}
